// SelfAttention_84052509983473
// MI455X (gfx1250) — matrix operand order repaired and checked by reading, not compiled, not run on hardware
//
#include <hip/hip_runtime.h>
#include <hip/hip_bf16.h>
#include <cstdint>
#include <cstddef>

// ---------------- types for CDNA5 WMMA ----------------
typedef _Float16 f16;
typedef __attribute__((ext_vector_type(16))) _Float16 v16h;
typedef __attribute__((ext_vector_type(8)))  _Float16 v8h;
typedef __attribute__((ext_vector_type(8)))  float    v8f;

union U16 { v16h v; v8h h[2]; f16 e[16]; };
union U8  { v8h  v; f16 e[8]; };

// ---------------- problem constants ----------------
static constexpr int   B_   = 16;
static constexpr int   C_   = 256;
static constexpr int   S_   = 1024;   // 32*32
static constexpr int   NH_  = 8;
static constexpr int   HD_  = 32;     // C/NH
static constexpr int   G_   = 8;
static constexpr int   CPG_ = 32;     // C/G
static constexpr float EPS_   = 1e-5f;
static constexpr float SCALE_ = 0.17677669529663687f;  // hd^-0.5

__device__ __forceinline__ v8f wmma_f16(v16h a, v16h b, v8f c) {
  // v_wmma_f32_16x16x32_f16: D = A(16x32) * B(32x16) + C
  return __builtin_amdgcn_wmma_f32_16x16x32_f16(false, a, false, b, (short)0, c,
                                                false, false);
}

// ---------------- kernel 1: fp32 -> f16 weight convert ----------------
__global__ void cvt_f16_kernel(const float* __restrict__ src,
                               f16* __restrict__ dst, int n) {
  int i = blockIdx.x * blockDim.x + threadIdx.x;
  if (i < n) dst[i] = (f16)src[i];
}

// ---------------- kernel 2: GroupNorm + transpose to [B,S,C] f16 ----------
__global__ void __launch_bounds__(256)
gn_kernel(const float* __restrict__ x, const float* __restrict__ gw,
          const float* __restrict__ gb, f16* __restrict__ xnT) {
  const int bg = blockIdx.x;           // B*G = 128 blocks
  const int b = bg >> 3, g = bg & 7;
  const int tid = threadIdx.x;
  const float* xb = x + ((size_t)b * C_ + (size_t)g * CPG_) * S_;  // 32 x 1024

  float sum = 0.f, sq = 0.f;
  for (int i = tid; i < CPG_ * S_; i += 256) {   // coalesced reads
    float v = xb[i];
    sum += v; sq += v * v;
  }
  __shared__ float s1[256], s2[256];
  s1[tid] = sum; s2[tid] = sq;
  __syncthreads();
  for (int st = 128; st > 0; st >>= 1) {
    if (tid < st) { s1[tid] += s1[tid + st]; s2[tid] += s2[tid + st]; }
    __syncthreads();
  }
  const float inv_n = 1.0f / (float)(CPG_ * S_);
  const float mean = s1[0] * inv_n;
  const float var  = s2[0] * inv_n - mean * mean;
  const float inv  = rsqrtf(var + EPS_);

  // normalize + affine + transpose store (consecutive tid -> consecutive c)
  for (int i = tid; i < CPG_ * S_; i += 256) {
    int cl = i & 31;
    int s  = i >> 5;
    int c  = g * CPG_ + cl;
    float v = xb[(size_t)cl * S_ + s];
    float y = (v - mean) * inv * gw[c] + gb[c];
    xnT[((size_t)b * S_ + s) * C_ + c] = (f16)y;
  }
}

// ---------------- kernel 3: QKV projection GEMM (f16 WMMA) ----------------
// one wave = 16(M) x 64(N) tile; blockIdx.y selects q/k/v
__global__ void __launch_bounds__(256)
qkv_kernel(const f16* __restrict__ xnT,
           const f16* __restrict__ wq16, const f16* __restrict__ wk16,
           const f16* __restrict__ wv16,
           const float* __restrict__ bq, const float* __restrict__ bk,
           const float* __restrict__ bv,
           f16* __restrict__ Qs, f16* __restrict__ Kf, f16* __restrict__ Vt) {
  const int z = blockIdx.y;
  const f16*   w    = (z == 0) ? wq16 : (z == 1) ? wk16 : wv16;
  const float* bias = (z == 0) ? bq   : (z == 1) ? bk   : bv;

  const int tid = threadIdx.x, wid = tid >> 5, lane = tid & 31;
  const int l = lane & 15, hl = lane >> 4;
  const int t  = blockIdx.x * 8 + wid;       // 4096 tiles total
  const int m0 = (t >> 2) * 16;              // row in flattened [B*S]
  const int n0 = (t & 3) * 64;               // output-channel block

  const f16* arow = xnT + (size_t)(m0 + l) * C_;
  v8f acc[4] = {v8f{}, v8f{}, v8f{}, v8f{}};

  for (int kb = 0; kb < C_; kb += 32) {
    U16 a;
    a.h[0] = *(const v8h*)(arow + kb + hl * 16);
    a.h[1] = *(const v8h*)(arow + kb + hl * 16 + 8);
#pragma unroll
    for (int nt = 0; nt < 4; ++nt) {
      v16h bm = *(const v16h*)(w + (size_t)(n0 + nt * 16 + l) * C_ + kb + hl * 16);
      acc[nt] = wmma_f16(a.v, bm, acc[nt]);
    }
  }

  const int b  = m0 >> 10;
  const int sb = (m0 & 1023) + hl * 8;
#pragma unroll
  for (int nt = 0; nt < 4; ++nt) {
    const int c_out = n0 + nt * 16 + l;
    const int h = c_out >> 5, d = c_out & 31;
    const float bi = bias[c_out];
    const size_t hb = (size_t)(b * NH_ + h);
    if (z == 0) {                 // Q: row-major [S,hd], scale folded in
#pragma unroll
      for (int r = 0; r < 8; ++r)
        Qs[(hb * S_ + sb + r) * HD_ + d] = (f16)((acc[nt][r] + bi) * SCALE_);
    } else if (z == 1) {          // K: row-major [S,hd]
#pragma unroll
      for (int r = 0; r < 8; ++r)
        Kf[(hb * S_ + sb + r) * HD_ + d] = (f16)(acc[nt][r] + bi);
    } else {                      // V: transposed [hd,S] -> contiguous store
      U8 pk;
#pragma unroll
      for (int r = 0; r < 8; ++r) pk.e[r] = (f16)(acc[nt][r] + bi);
      *(v8h*)(Vt + (hb * HD_ + d) * S_ + sb) = pk.v;
    }
  }
}

// ---------------- kernel 4: fused flash attention --------------------------
// wave = 16 query rows of one (b,h); 32 iterations over 32-key blocks.
// Per-row softmax bias term in the reference is constant over the softmax
// axis -> drops out exactly; omitted.
__global__ void __launch_bounds__(256)
attn_kernel(const f16* __restrict__ Qs, const f16* __restrict__ Kf,
            const f16* __restrict__ Vt, f16* __restrict__ AO) {
  const int tid = threadIdx.x, wid = tid >> 5, lane = tid & 31;
  const int l = lane & 15, hl = lane >> 4;
  const int gw = blockIdx.x * 8 + wid;   // 8192 waves total
  const int bh = gw >> 6;                // (b*NH + h)
  const int i0 = (gw & 63) * 16;         // query row block

  // A operand: Q tile 16x32, loaded once (K-dim == hd)
  U16 aq;
  const f16* qrow = Qs + ((size_t)bh * S_ + i0 + l) * HD_;
  aq.h[0] = *(const v8h*)(qrow + hl * 16);
  aq.h[1] = *(const v8h*)(qrow + hl * 16 + 8);

  float mr[8], lr[8];
#pragma unroll
  for (int r = 0; r < 8; ++r) { mr[r] = -3.0e38f; lr[r] = 0.f; }
  v8f o0 = {}, o1 = {};

  __shared__ f16 PB[8][16][40];          // per-wave D->A relayout bounce

  for (int jb = 0; jb < S_; jb += 32) {
    // scores S = Q K^T  (two 16-col tiles)
    const f16* krow = Kf + ((size_t)bh * S_ + jb) * HD_;
    v16h b0 = *(const v16h*)(krow + (size_t)l * HD_ + hl * 16);
    v16h b1 = *(const v16h*)(krow + (size_t)(16 + l) * HD_ + hl * 16);
    v8f zc = {};
    v8f s0 = wmma_f16(aq.v, b0, zc);
    v8f s1 = wmma_f16(aq.v, b1, zc);

    // online softmax; D layout: VGPR r = row (r + 8*hl), col l
#pragma unroll
    for (int r = 0; r < 8; ++r) {
      float x0 = s0[r], x1 = s1[r];
      float bm = fmaxf(x0, x1);
      bm = fmaxf(bm, __shfl_xor(bm, 8, 32));
      bm = fmaxf(bm, __shfl_xor(bm, 4, 32));
      bm = fmaxf(bm, __shfl_xor(bm, 2, 32));
      bm = fmaxf(bm, __shfl_xor(bm, 1, 32));
      float mn = fmaxf(mr[r], bm);
      float alpha = __expf(mr[r] - mn);
      float p0 = __expf(x0 - mn);
      float p1 = __expf(x1 - mn);
      float bl = p0 + p1;
      bl += __shfl_xor(bl, 8, 32);
      bl += __shfl_xor(bl, 4, 32);
      bl += __shfl_xor(bl, 2, 32);
      bl += __shfl_xor(bl, 1, 32);
      lr[r] = lr[r] * alpha + bl;
      mr[r] = mn;
      o0[r] *= alpha;
      o1[r] *= alpha;
      const int m = r + 8 * hl;
      PB[wid][m][l]      = (f16)p0;
      PB[wid][m][16 + l] = (f16)p1;
    }

    // P (A layout from LDS) x V (B layout from V^T, contiguous)
    U16 ap;
    ap.h[0] = *(const v8h*)&PB[wid][l][hl * 16];
    ap.h[1] = *(const v8h*)&PB[wid][l][hl * 16 + 8];
    const f16* vbase = Vt + (size_t)bh * HD_ * S_ + jb + hl * 16;
    v16h bv0 = *(const v16h*)(vbase + (size_t)l * S_);
    v16h bv1 = *(const v16h*)(vbase + (size_t)(16 + l) * S_);
    o0 = wmma_f16(ap.v, bv0, o0);
    o1 = wmma_f16(ap.v, bv1, o1);
  }

  // normalize and store heads-concat layout [B,S,C] f16
  const int h = bh & 7, b = bh >> 3;
#pragma unroll
  for (int r = 0; r < 8; ++r) {
    const int s = i0 + r + 8 * hl;
    const float inv = 1.0f / lr[r];
    f16* orow = AO + ((size_t)b * S_ + s) * C_ + h * HD_;
    orow[l]      = (f16)(o0[r] * inv);
    orow[16 + l] = (f16)(o1[r] * inv);
  }
}

// ---------------- kernel 5: output projection, store [B,C,S] fp32 ---------
__global__ void __launch_bounds__(256)
oproj_kernel(const f16* __restrict__ AO, const f16* __restrict__ wo16,
             const float* __restrict__ bo, float* __restrict__ out) {
  const int tid = threadIdx.x, wid = tid >> 5, lane = tid & 31;
  const int l = lane & 15, hl = lane >> 4;
  const int t  = blockIdx.x * 8 + wid;
  const int m0 = (t >> 2) * 16;
  const int n0 = (t & 3) * 64;

  const f16* arow = AO + (size_t)(m0 + l) * C_;
  v8f acc[4] = {v8f{}, v8f{}, v8f{}, v8f{}};

  for (int kb = 0; kb < C_; kb += 32) {
    U16 a;
    a.h[0] = *(const v8h*)(arow + kb + hl * 16);
    a.h[1] = *(const v8h*)(arow + kb + hl * 16 + 8);
#pragma unroll
    for (int nt = 0; nt < 4; ++nt) {
      v16h bm = *(const v16h*)(wo16 + (size_t)(n0 + nt * 16 + l) * C_ + kb + hl * 16);
      acc[nt] = wmma_f16(a.v, bm, acc[nt]);
    }
  }

  const int b  = m0 >> 10;
  const int sb = (m0 & 1023) + hl * 8;
#pragma unroll
  for (int nt = 0; nt < 4; ++nt) {
    const int c_out = n0 + nt * 16 + l;
    const float bi = bo[c_out];
    float* orow = out + ((size_t)b * C_ + c_out) * S_ + sb;  // contiguous in s
#pragma unroll
    for (int r = 0; r < 8; ++r) orow[r] = acc[nt][r] + bi;
  }
}

// ---------------- host-side launcher ----------------
extern "C" void kernel_launch(void* const* d_in, const int* in_sizes, int n_in,
                              void* d_out, int out_size, void* d_ws, size_t ws_size,
                              hipStream_t stream) {
  (void)in_sizes; (void)n_in; (void)out_size; (void)ws_size;
  const float* x    = (const float*)d_in[0];
  const float* wq   = (const float*)d_in[1];
  const float* bq   = (const float*)d_in[2];
  const float* wk   = (const float*)d_in[3];
  const float* bk   = (const float*)d_in[4];
  const float* wv   = (const float*)d_in[5];
  const float* bv   = (const float*)d_in[6];
  const float* wo   = (const float*)d_in[7];
  const float* bo   = (const float*)d_in[8];
  const float* gn_w = (const float*)d_in[9];
  const float* gn_b = (const float*)d_in[10];
  float* out = (float*)d_out;

  // workspace layout (bytes)
  char* ws = (char*)d_ws;
  const size_t WSZ = (size_t)C_ * C_ * sizeof(f16);            // 128 KB per weight
  f16* wq16 = (f16*)(ws + 0 * WSZ);
  f16* wk16 = (f16*)(ws + 1 * WSZ);
  f16* wv16 = (f16*)(ws + 2 * WSZ);
  f16* wo16 = (f16*)(ws + 3 * WSZ);
  const size_t TSZ = (size_t)B_ * S_ * C_ * sizeof(f16);       // 8 MB per tensor
  f16* xnT = (f16*)(ws + 4 * WSZ);                              // reused as AO
  f16* Qs  = (f16*)(ws + 4 * WSZ + 1 * TSZ);
  f16* Kf  = (f16*)(ws + 4 * WSZ + 2 * TSZ);
  f16* Vt  = (f16*)(ws + 4 * WSZ + 3 * TSZ);
  f16* AO  = xnT;  // alias: xnT fully consumed before attention writes

  const int nw = C_ * C_;
  cvt_f16_kernel<<<(nw + 255) / 256, 256, 0, stream>>>(wq, wq16, nw);
  cvt_f16_kernel<<<(nw + 255) / 256, 256, 0, stream>>>(wk, wk16, nw);
  cvt_f16_kernel<<<(nw + 255) / 256, 256, 0, stream>>>(wv, wv16, nw);
  cvt_f16_kernel<<<(nw + 255) / 256, 256, 0, stream>>>(wo, wo16, nw);

  gn_kernel<<<B_ * G_, 256, 0, stream>>>(x, gn_w, gn_b, xnT);

  qkv_kernel<<<dim3(512, 3), 256, 0, stream>>>(xnT, wq16, wk16, wv16,
                                               bq, bk, bv, Qs, Kf, Vt);

  attn_kernel<<<1024, 256, 0, stream>>>(Qs, Kf, Vt, AO);

  oproj_kernel<<<512, 256, 0, stream>>>(AO, wo16, bo, out);
}
